// Attention_45715631899552
// MI455X (gfx1250) — compile-verified, matrix operand order checked by reading, not run on hardware
//
#include <hip/hip_runtime.h>
#include <hip/hip_bf16.h>

typedef __attribute__((ext_vector_type(16))) _Float16 v16h;
typedef __attribute__((ext_vector_type(8)))  _Float16 v8h;
typedef __attribute__((ext_vector_type(8)))  float    v8f;

#define BB   16
#define NN   800
#define NCC  128
#define LL   928          // N + NC
#define CC   1024
#define HH   16
#define HD   64
#define SCALE 0.125f      // 1/sqrt(64)
#define SP   944          // padded LDS score row stride (f32 elements)

// ---------------------------------------------------------------------------
// Fragment loader for V_WMMA_F32_16X16X32_F16 (A and B share the layout,
// lane = row(mod 16); lanes 16-31 take the +8 K-half; see CDNA5 ISA):
//   elements 0..7  = K[koff .. koff+7]       (koff = (lane>>4)*8)
//   elements 8..15 = K[16+koff .. 16+koff+7]
// -> two contiguous 16-byte loads per lane.
// ---------------------------------------------------------------------------
__device__ __forceinline__ v16h load_frag(const _Float16* base, int ldm) {
  const int lane = threadIdx.x & 31;
  const int r    = lane & 15;
  const int koff = (lane >> 4) << 3;
  const _Float16* p = base + (size_t)r * ldm + koff;
  v8h lo = *(const v8h*)(p);
  v8h hi = *(const v8h*)(p + 16);
  return __builtin_shufflevector(lo, hi, 0,1,2,3,4,5,6,7,8,9,10,11,12,13,14,15);
}

__device__ __forceinline__ v8f wmma32(v16h a, v16h b, v8f c) {
  // D = A(16x32 f16) x B(32x16 f16) + C(16x16 f32)
  return __builtin_amdgcn_wmma_f32_16x16x32_f16(false, a, false, b,
                                                (short)0, c, false, false);
}

// ---------------------------------------------------------------------------
// Prep: f16 staging of concatenated K/V inputs and weights
// ---------------------------------------------------------------------------
__global__ void k_prep_kin(const float* __restrict__ x, const float* __restrict__ xpos,
                           const float* __restrict__ k_t, _Float16* __restrict__ kin) {
  size_t i = (size_t)blockIdx.x * blockDim.x + threadIdx.x;
  const size_t total = (size_t)BB * LL * CC;
  if (i >= total) return;
  size_t c = i % CC;
  size_t r = (i / CC) % LL;
  size_t b = i / ((size_t)LL * CC);
  float v;
  if (r < NN) { size_t j = (b * NN + r) * CC + c; v = x[j] + xpos[j]; }
  else        { size_t j = (b * NCC + (r - NN)) * CC + c; v = k_t[j]; }
  kin[i] = (_Float16)v;
}

__global__ void k_prep_vin(const float* __restrict__ x, const float* __restrict__ v_t,
                           _Float16* __restrict__ vin) {
  size_t i = (size_t)blockIdx.x * blockDim.x + threadIdx.x;
  const size_t total = (size_t)BB * LL * CC;
  if (i >= total) return;
  size_t c = i % CC;
  size_t r = (i / CC) % LL;
  size_t b = i / ((size_t)LL * CC);
  float v;
  if (r < NN) v = x[(b * NN + r) * CC + c];
  else        v = v_t[(b * NCC + (r - NN)) * CC + c];
  vin[i] = (_Float16)v;
}

__global__ void k_f32_to_f16(const float* __restrict__ s, _Float16* __restrict__ d, int n) {
  int i = blockIdx.x * blockDim.x + threadIdx.x;
  if (i < n) d[i] = (_Float16)s[i];
}

// ---------------------------------------------------------------------------
// Projection GEMM: Y[b] = X[b] @ W^T   (f16 in, f16 out)
// Register-blocked: each wave computes a 16(M) x 64(N) strip = 4 accumulator
// tiles, reusing the A fragment 4x per k-step (4x arithmetic intensity).
// The 64-column strip is exactly one head, so the vtrans=1 store to
// Y[B][H][HD][LL] is 4 contiguous-row writes.
// ---------------------------------------------------------------------------
__global__ void k_gemm_proj(const _Float16* __restrict__ X, const _Float16* __restrict__ W,
                            _Float16* __restrict__ Y, size_t sX, size_t sY, int vtrans) {
  const int wave = threadIdx.x >> 5;
  const int lane = threadIdx.x & 31;
  const int m0 = blockIdx.x * 16;
  const int n0 = (blockIdx.y * (blockDim.x >> 5) + wave) * 64;  // 64-col strip
  const int b  = blockIdx.z;

  const _Float16* Xb = X + (size_t)b * sX + (size_t)m0 * CC;
  const _Float16* Wb = W + (size_t)n0 * CC;

  v8f acc0 = {}, acc1 = {}, acc2 = {}, acc3 = {};
  #pragma unroll 2
  for (int k = 0; k < CC; k += 32) {
    v16h a  = load_frag(Xb + k, CC);                 // A reused 4x
    v16h w0 = load_frag(Wb + (size_t) 0 * CC + k, CC);
    v16h w1 = load_frag(Wb + (size_t)16 * CC + k, CC);
    v16h w2 = load_frag(Wb + (size_t)32 * CC + k, CC);
    v16h w3 = load_frag(Wb + (size_t)48 * CC + k, CC);
    acc0 = wmma32(a, w0, acc0);
    acc1 = wmma32(a, w1, acc1);
    acc2 = wmma32(a, w2, acc2);
    acc3 = wmma32(a, w3, acc3);
  }

  const int n     = lane & 15;
  const int rbase = (lane >> 4) << 3;     // D: VGPR r -> rows r / r+8
  if (!vtrans) {
    _Float16* Yp = Y + (size_t)b * sY + (size_t)m0 * CC + n0 + n;
    #pragma unroll
    for (int r = 0; r < 8; ++r) {
      const size_t ro = (size_t)(rbase + r) * CC;
      Yp[ro +  0] = (_Float16)acc0[r];
      Yp[ro + 16] = (_Float16)acc1[r];
      Yp[ro + 32] = (_Float16)acc2[r];
      Yp[ro + 48] = (_Float16)acc3[r];
    }
  } else {
    // strip covers head h = n0/64; tile t covers d = t*16 + n
    const int h = n0 >> 6;
    _Float16* Yp = Y + (size_t)b * sY + (size_t)h * HD * LL + m0 + rbase;
    #pragma unroll
    for (int r = 0; r < 8; ++r) {
      Yp[(size_t)( 0 + n) * LL + r] = (_Float16)acc0[r];
      Yp[(size_t)(16 + n) * LL + r] = (_Float16)acc1[r];
      Yp[(size_t)(32 + n) * LL + r] = (_Float16)acc2[r];
      Yp[(size_t)(48 + n) * LL + r] = (_Float16)acc3[r];
    }
  }
}

// ---------------------------------------------------------------------------
// Attention: one wave per (b, h, 16-query tile).
//   S = (Q K^T) * scale  -> LDS f32 [16][SP]
//   softmax (unnormalized exp -> f16 in-place alias, row sums -> LDS)
//   O = P V  (V pre-transposed per head), normalize by row sum at store
// ---------------------------------------------------------------------------
__global__ void k_attn(const _Float16* __restrict__ Q, const _Float16* __restrict__ K,
                       const _Float16* __restrict__ VT, _Float16* __restrict__ O) {
  __shared__ __align__(16) float smem[16 * SP + 16];
  float*    rowsum = smem + 16 * SP;
  _Float16* P      = (_Float16*)smem;     // f16 alias, identical flat indexing

  const int lane = threadIdx.x & 31;
  const int q0 = blockIdx.x * 16;
  const int h  = blockIdx.y;
  const int b  = blockIdx.z;

  const _Float16* Qb = Q + ((size_t)b * NN + q0) * CC + h * HD;
  const _Float16* Kb = K + (size_t)b * LL * CC + h * HD;
  const _Float16* Vb = VT + ((size_t)b * HH + h) * (size_t)HD * LL;

  const v16h a0 = load_frag(Qb + 0,  CC);   // q rows, d = 0..31
  const v16h a1 = load_frag(Qb + 32, CC);   // q rows, d = 32..63

  const int n     = lane & 15;
  const int rbase = (lane >> 4) << 3;

  // ---- pass 1: scaled scores to LDS --------------------------------------
  for (int j0 = 0; j0 < LL; j0 += 16) {
    v16h b0 = load_frag(Kb + (size_t)j0 * CC + 0,  CC);   // B[k=d, n=j] = K[j,d]
    v16h b1 = load_frag(Kb + (size_t)j0 * CC + 32, CC);
    v8f s = {};
    s = wmma32(a0, b0, s);
    s = wmma32(a1, b1, s);
    #pragma unroll
    for (int r = 0; r < 8; ++r)
      smem[(rbase + r) * SP + j0 + n] = s[r] * SCALE;
  }
  __syncthreads();   // single-wave WG: S_NOP barrier, acts as compiler fence

  // ---- pass 2: row softmax (unnormalized), f16 exp written in place -------
  for (int r = 0; r < 16; ++r) {
    const float* row = smem + r * SP;
    float m = -1e30f;
    for (int c = lane; c < LL; c += 32) m = fmaxf(m, row[c]);
    #pragma unroll
    for (int o = 16; o > 0; o >>= 1) m = fmaxf(m, __shfl_xor(m, o, 32));
    float sum = 0.f;
    volatile _Float16* prow = (volatile _Float16*)(P + r * SP);
    for (int c = lane; c < LL; c += 32) {
      float e = __expf(row[c] - m);     // read f32 @4f precedes write f16 @2f
      sum += e;
      prow[c] = (_Float16)e;
    }
    #pragma unroll
    for (int o = 16; o > 0; o >>= 1) sum += __shfl_xor(sum, o, 32);
    if (lane == 0) rowsum[r] = sum;
  }
  __syncthreads();

  // ---- pass 3: O = P @ V, 4 d-tiles of 16 ---------------------------------
  v8f o0 = {}, o1 = {}, o2 = {}, o3 = {};
  for (int j0 = 0; j0 < LL; j0 += 32) {
    v16h ap = load_frag(P + j0, SP);                       // A: q rows, K = j
    o0 = wmma32(ap, load_frag(Vb + (size_t) 0 * LL + j0, LL), o0);
    o1 = wmma32(ap, load_frag(Vb + (size_t)16 * LL + j0, LL), o1);
    o2 = wmma32(ap, load_frag(Vb + (size_t)32 * LL + j0, LL), o2);
    o3 = wmma32(ap, load_frag(Vb + (size_t)48 * LL + j0, LL), o3);
  }

  _Float16* Ob = O + ((size_t)b * NN + q0) * CC + h * HD;
  #pragma unroll
  for (int r = 0; r < 8; ++r) {
    const int row = rbase + r;
    const float rin = 1.0f / rowsum[row];
    Ob[(size_t)row * CC +  0 + n] = (_Float16)(o0[r] * rin);
    Ob[(size_t)row * CC + 16 + n] = (_Float16)(o1[r] * rin);
    Ob[(size_t)row * CC + 32 + n] = (_Float16)(o2[r] * rin);
    Ob[(size_t)row * CC + 48 + n] = (_Float16)(o3[r] * rin);
  }
}

// ---------------------------------------------------------------------------
// Final projection: out = attn @ Wp^T + bp   (f32 output)
// Same 16x64 register blocking as k_gemm_proj.
// ---------------------------------------------------------------------------
__global__ void k_gemm_final(const _Float16* __restrict__ X, const _Float16* __restrict__ W,
                             const float* __restrict__ bias, float* __restrict__ Y) {
  const int wave = threadIdx.x >> 5;
  const int lane = threadIdx.x & 31;
  const int m0 = blockIdx.x * 16;
  const int n0 = (blockIdx.y * (blockDim.x >> 5) + wave) * 64;
  const int b  = blockIdx.z;

  const _Float16* Xb = X + ((size_t)b * NN + m0) * CC;
  const _Float16* Wb = W + (size_t)n0 * CC;

  v8f acc0 = {}, acc1 = {}, acc2 = {}, acc3 = {};
  #pragma unroll 2
  for (int k = 0; k < CC; k += 32) {
    v16h a  = load_frag(Xb + k, CC);
    acc0 = wmma32(a, load_frag(Wb + (size_t) 0 * CC + k, CC), acc0);
    acc1 = wmma32(a, load_frag(Wb + (size_t)16 * CC + k, CC), acc1);
    acc2 = wmma32(a, load_frag(Wb + (size_t)32 * CC + k, CC), acc2);
    acc3 = wmma32(a, load_frag(Wb + (size_t)48 * CC + k, CC), acc3);
  }

  const int n     = lane & 15;
  const int rbase = (lane >> 4) << 3;
  const float b0v = bias[n0 +  0 + n];
  const float b1v = bias[n0 + 16 + n];
  const float b2v = bias[n0 + 32 + n];
  const float b3v = bias[n0 + 48 + n];
  float* Yp = Y + ((size_t)b * NN + m0) * CC + n0 + n;
  #pragma unroll
  for (int r = 0; r < 8; ++r) {
    const size_t ro = (size_t)(rbase + r) * CC;
    Yp[ro +  0] = acc0[r] + b0v;
    Yp[ro + 16] = acc1[r] + b1v;
    Yp[ro + 32] = acc2[r] + b2v;
    Yp[ro + 48] = acc3[r] + b3v;
  }
}

// ---------------------------------------------------------------------------
extern "C" void kernel_launch(void* const* d_in, const int* in_sizes, int n_in,
                              void* d_out, int out_size, void* d_ws, size_t ws_size,
                              hipStream_t stream) {
  (void)in_sizes; (void)n_in; (void)out_size; (void)ws_size;
  const float* x    = (const float*)d_in[0];
  const float* xpos = (const float*)d_in[1];
  const float* k_t  = (const float*)d_in[2];
  const float* v_t  = (const float*)d_in[3];
  const float* Wq   = (const float*)d_in[4];
  const float* Wk   = (const float*)d_in[5];
  const float* Wv   = (const float*)d_in[6];
  const float* Wp   = (const float*)d_in[7];
  const float* bp   = (const float*)d_in[8];
  float* out = (float*)d_out;

  char* ws = (char*)d_ws;
  size_t off = 0;
  auto carve = [&](size_t bytes) -> char* {
    char* p = ws + off;
    off += (bytes + 255) & ~(size_t)255;
    return p;
  };
  const size_t nBL = (size_t)BB * LL * CC;   // 15,204,352
  const size_t nBN = (size_t)BB * NN * CC;   // 13,107,200
  _Float16* kin = (_Float16*)carve(nBL * 2);
  _Float16* vin = (_Float16*)carve(nBL * 2);
  _Float16* wqh = (_Float16*)carve((size_t)CC * CC * 2);
  _Float16* wkh = (_Float16*)carve((size_t)CC * CC * 2);
  _Float16* wvh = (_Float16*)carve((size_t)CC * CC * 2);
  _Float16* wph = (_Float16*)carve((size_t)CC * CC * 2);
  _Float16* qb  = (_Float16*)carve(nBN * 2);
  _Float16* kb  = (_Float16*)carve(nBL * 2);
  _Float16* vtb = (_Float16*)carve(nBL * 2);  // [B][H][HD][LL]
  _Float16* ab  = (_Float16*)carve(nBN * 2);  // attention output, f16

  // stage f16 inputs
  const int nw = CC * CC;
  k_f32_to_f16<<<(nw + 255) / 256, 256, 0, stream>>>(Wq, wqh, nw);
  k_f32_to_f16<<<(nw + 255) / 256, 256, 0, stream>>>(Wk, wkh, nw);
  k_f32_to_f16<<<(nw + 255) / 256, 256, 0, stream>>>(Wv, wvh, nw);
  k_f32_to_f16<<<(nw + 255) / 256, 256, 0, stream>>>(Wp, wph, nw);
  k_prep_kin<<<(unsigned)((nBL + 255) / 256), 256, 0, stream>>>(x, xpos, k_t, kin);
  k_prep_vin<<<(unsigned)((nBL + 255) / 256), 256, 0, stream>>>(x, v_t, vin);

  // projections: q (first 800 rows of kin), k (928 rows), v (928 rows, transposed)
  // grid.y: 1024 cols / 64 per wave / 4 waves per block = 4
  k_gemm_proj<<<dim3(NN / 16, CC / 64 / 4, BB), 128, 0, stream>>>(
      kin, wqh, qb, (size_t)LL * CC, (size_t)NN * CC, 0);
  k_gemm_proj<<<dim3(LL / 16, CC / 64 / 4, BB), 128, 0, stream>>>(
      kin, wkh, kb, (size_t)LL * CC, (size_t)LL * CC, 0);
  k_gemm_proj<<<dim3(LL / 16, CC / 64 / 4, BB), 128, 0, stream>>>(
      vin, wvh, vtb, (size_t)LL * CC, (size_t)LL * CC, 1);

  // attention
  k_attn<<<dim3(NN / 16, HH, BB), 32, 0, stream>>>(qb, kb, vtb, ab);

  // output projection (+bias) to f32
  k_gemm_final<<<dim3(NN / 16, CC / 64 / 4, BB), 128, 0, stream>>>(ab, wph, bp, out);
}
